// PositionalEncoding_34591666602099
// MI455X (gfx1250) — matrix operand order repaired and checked by reading, not compiled, not run on hardware
//
#include <hip/hip_runtime.h>

// ---------------------------------------------------------------------------
// CDNA5 (gfx1250) wave32 WMMA implementation of a PointNet++ SA module:
//   two (ball_query -> group -> sharedMLP(6->32->64->128)+BN+ReLU -> maxpool)
//   branches, concat -> [16384,256] @ [256,256] + bias.
// All matrix math runs on v_wmma_f32_16x16x32_f16 (f16 in, f32 accumulate),
// BN folded into f16 weights held in LDS in a B-fragment-friendly transposed
// layout; activations ping through a per-wave LDS tile laid out for direct
// A-fragment b128 loads.
// ---------------------------------------------------------------------------

typedef __attribute__((ext_vector_type(16))) _Float16 v16h;
typedef __attribute__((ext_vector_type(8)))  _Float16 v8h;
typedef __attribute__((ext_vector_type(8)))  float    v8f;

#define BN_EPS 1e-5f

struct MlpParams {
  const float* W[3];
  const float* b[3];
  const float* g[3];
  const float* be[3];
  const float* rm[3];
  const float* rv[3];
};

// A fragment (16x32 f16, row-major source, strideH halves per row).
// lane = lo + 16*hi; caller passes p = base + row*strideH + kbase + hi*8.
// lanes 0-15 hold K = kbase+{0..7, 16..23}; lanes 16-31 K = kbase+{8..15, 24..31}.
__device__ __forceinline__ v16h load_a16(const _Float16* p) {
  union { v16h v; v8h h[2]; } u;
  u.h[0] = *(const v8h*)(p);
  u.h[1] = *(const v8h*)(p + 16);
  return u.v;
}

// B fragment (32x16 f16) from TRANSPOSED weights Wt[n][k].
// caller passes p = wt + n*strideH + kbase + hi*8; K = {hi*8..hi*8+7, 16+hi*8..16+hi*8+7}.
__device__ __forceinline__ v16h load_b16(const _Float16* p) {
  union { v16h v; v8h h[2]; } u;
  u.h[0] = *(const v8h*)(p);
  u.h[1] = *(const v8h*)(p + 16);
  return u.v;
}

__device__ __forceinline__ v8f wmma_f16(v16h a, v16h b, v8f c) {
  return __builtin_amdgcn_wmma_f32_16x16x32_f16(false, a, false, b,
                                                (short)0, c, false, false);
}

// ---------------------------------------------------------------------------
// Branch kernel: one wave per query. NS = 32 or 64 neighbors.
// ---------------------------------------------------------------------------
template <int NS>
__global__ __launch_bounds__(128, 1)
void pointnet_branch_kernel(const float* __restrict__ pts, MlpParams P,
                            float r2, _Float16* __restrict__ fbuf, int colOff) {
  constexpr int MT = NS / 16;   // M tiles per query
  constexpr int NPTS = 2048;
  constexpr int SA = 72;        // activation row stride (halves), 144B = 16B-aligned

  __shared__ _Float16 sW1[32 * 32];       // [n][k], k padded 6->32, BN-folded
  __shared__ _Float16 sW2[64 * 32];       // [n][k]
  __shared__ _Float16 sW3[128 * 64];      // [n][k]
  __shared__ float    sB1[32], sB2[64], sB3[128];
  __shared__ _Float16 sAct[4][64 * SA];   // per-wave activation tile
  __shared__ int      sIdx[4][64];        // per-wave neighbor list

  const int tid = threadIdx.x;

  // ---- fold BN into transposed f16 weights (once per block) ----
  for (int t = tid; t < 32 * 32; t += 128) {
    int n = t >> 5, k = t & 31;
    float s = P.g[0][n] * rsqrtf(P.rv[0][n] + BN_EPS);
    sW1[n * 32 + k] = (k < 6) ? (_Float16)(P.W[0][k * 32 + n] * s) : (_Float16)0.f;
  }
  for (int t = tid; t < 64 * 32; t += 128) {
    int n = t >> 5, k = t & 31;
    float s = P.g[1][n] * rsqrtf(P.rv[1][n] + BN_EPS);
    sW2[n * 32 + k] = (_Float16)(P.W[1][k * 64 + n] * s);
  }
  for (int t = tid; t < 128 * 64; t += 128) {
    int n = t >> 6, k = t & 63;
    float s = P.g[2][n] * rsqrtf(P.rv[2][n] + BN_EPS);
    sW3[n * 64 + k] = (_Float16)(P.W[2][k * 128 + n] * s);
  }
  for (int t = tid; t < 32; t += 128) {
    float s = P.g[0][t] * rsqrtf(P.rv[0][t] + BN_EPS);
    sB1[t] = (P.b[0][t] - P.rm[0][t]) * s + P.be[0][t];
  }
  for (int t = tid; t < 64; t += 128) {
    float s = P.g[1][t] * rsqrtf(P.rv[1][t] + BN_EPS);
    sB2[t] = (P.b[1][t] - P.rm[1][t]) * s + P.be[1][t];
  }
  for (int t = tid; t < 128; t += 128) {
    float s = P.g[2][t] * rsqrtf(P.rv[2][t] + BN_EPS);
    sB3[t] = (P.b[2][t] - P.rm[2][t]) * s + P.be[2][t];
  }
  __syncthreads();

  const int w    = tid >> 5;
  const int lane = tid & 31;
  const int lo   = lane & 15;
  const int hi   = lane >> 4;
  _Float16* act  = sAct[w];
  int*      idxl = sIdx[w];

  const int nWaves = gridDim.x * 4;
  for (int q = blockIdx.x * 4 + w; q < 8 * NPTS; q += nWaves) {
    const int bb = q >> 11;
    const int m  = q & 2047;
    const float* base = pts + (size_t)bb * (NPTS * 3);
    const float cx = base[m * 3 + 0];
    const float cy = base[m * 3 + 1];
    const float cz = base[m * 3 + 2];

    // ---- ball query: first NS ascending indices with d2 < r2 ----
    int cnt = 0;
    for (int s0 = 0; s0 < NPTS && cnt < NS; s0 += 32) {
      int n = s0 + lane;
      float dx = base[n * 3 + 0] - cx;
      float dy = base[n * 3 + 1] - cy;
      float dz = base[n * 3 + 2] - cz;
      bool hit = (dx * dx + dy * dy + dz * dz) < r2;
      unsigned mask = (unsigned)__ballot(hit);
      if (hit) {
        int pos = cnt + __popc(mask & ((1u << lane) - 1u));
        if (pos < NS) idxl[pos] = n;
      }
      cnt += __popc(mask);
    }
    if (cnt > NS) cnt = NS;
    __asm__ volatile("s_wait_dscnt 0" ::: "memory");
    int first = (cnt > 0) ? idxl[0] : 0;
    for (int j = cnt + lane; j < NS; j += 32) idxl[j] = first;

    // ---- group: rows [NS][32], cols 0..5 = (p-c, p), 6..31 = 0 ----
    for (int j = lane; j < NS; j += 32) {
      int id = idxl[j];
      float px = base[id * 3 + 0], py = base[id * 3 + 1], pz = base[id * 3 + 2];
      v8h row0 = { (_Float16)(px - cx), (_Float16)(py - cy), (_Float16)(pz - cz),
                   (_Float16)px, (_Float16)py, (_Float16)pz,
                   (_Float16)0.f, (_Float16)0.f };
      v8h z = {};
      *(v8h*)(act + j * SA + 0)  = row0;
      *(v8h*)(act + j * SA + 8)  = z;
      *(v8h*)(act + j * SA + 16) = z;
      *(v8h*)(act + j * SA + 24) = z;
    }

    // ---- layer 1: [NS,32] x [32,32]  (K padded from 6) ----
    {
      v16h a[MT];
#pragma unroll
      for (int mt = 0; mt < MT; ++mt)
        a[mt] = load_a16(act + (mt * 16 + lo) * SA + hi * 8);
#pragma unroll
      for (int nt = 0; nt < 2; ++nt) {
        v16h bm = load_b16(sW1 + (nt * 16 + lo) * 32 + hi * 8);
        float bias = sB1[nt * 16 + lo];
#pragma unroll
        for (int mt = 0; mt < MT; ++mt) {
          v8f c = {};
          c = wmma_f16(a[mt], bm, c);
#pragma unroll
          for (int r = 0; r < 8; ++r) {
            float v = fmaxf(c[r] + bias, 0.f);
            act[(mt * 16 + r + hi * 8) * SA + nt * 16 + lo] = (_Float16)v;
          }
        }
      }
    }

    // ---- layer 2: [NS,32] x [32,64] ----
    {
      v16h a[MT];
#pragma unroll
      for (int mt = 0; mt < MT; ++mt)
        a[mt] = load_a16(act + (mt * 16 + lo) * SA + hi * 8);
#pragma unroll
      for (int nt = 0; nt < 4; ++nt) {
        v16h bm = load_b16(sW2 + (nt * 16 + lo) * 32 + hi * 8);
        float bias = sB2[nt * 16 + lo];
#pragma unroll
        for (int mt = 0; mt < MT; ++mt) {
          v8f c = {};
          c = wmma_f16(a[mt], bm, c);
#pragma unroll
          for (int r = 0; r < 8; ++r) {
            float v = fmaxf(c[r] + bias, 0.f);
            act[(mt * 16 + r + hi * 8) * SA + nt * 16 + lo] = (_Float16)v;
          }
        }
      }
    }

    // ---- layer 3: [NS,64] x [64,128] fused with max-pool over neighbors ----
    {
      v16h a0[MT], a1[MT];
#pragma unroll
      for (int mt = 0; mt < MT; ++mt) {
        a0[mt] = load_a16(act + (mt * 16 + lo) * SA + hi * 8);
        a1[mt] = load_a16(act + (mt * 16 + lo) * SA + 32 + hi * 8);
      }
#pragma unroll
      for (int nt = 0; nt < 8; ++nt) {
        v16h b0 = load_b16(sW3 + (nt * 16 + lo) * 64 + hi * 8);
        v16h b1 = load_b16(sW3 + (nt * 16 + lo) * 64 + 32 + hi * 8);
        float bias = sB3[nt * 16 + lo];
        float pmax = 0.f;  // ReLU outputs are >= 0, so 0 is the pool identity
#pragma unroll
        for (int mt = 0; mt < MT; ++mt) {
          v8f c = {};
          c = wmma_f16(a0[mt], b0, c);
          c = wmma_f16(a1[mt], b1, c);
#pragma unroll
          for (int r = 0; r < 8; ++r)
            pmax = fmaxf(pmax, c[r] + bias);
        }
        // merge the two M-halves (lanes L and L^16 cover disjoint rows, same N)
        pmax = fmaxf(pmax, __shfl_xor(pmax, 16, 32));
        if (hi == 0)
          fbuf[(size_t)q * 256 + colOff + nt * 16 + lo] = (_Float16)pmax;
      }
    }
  }
}

// ---------------------------------------------------------------------------
// w3 [256,256] f32 -> WT [n][k] f16 (B-fragment friendly)
// ---------------------------------------------------------------------------
__global__ void transpose_w3_kernel(const float* __restrict__ w3,
                                    _Float16* __restrict__ WT) {
  int idx = blockIdx.x * blockDim.x + threadIdx.x;  // 65536
  int n = idx & 255, k = idx >> 8;
  WT[n * 256 + k] = (_Float16)w3[k * 256 + n];
}

// ---------------------------------------------------------------------------
// Final GEMM: [rows,256](f16) x [256,256](f16) + b3 -> out f32.
// One wave = 16-row x 64-col tile (4 n-tiles), 8 K-steps.
// ---------------------------------------------------------------------------
__global__ __launch_bounds__(256, 1)
void final_gemm_kernel(const _Float16* __restrict__ A,
                       const _Float16* __restrict__ WT,
                       const float* __restrict__ b3,
                       float* __restrict__ out, int rows) {
  const int tid  = threadIdx.x;
  const int lane = tid & 31;
  const int lo   = lane & 15, hi = lane >> 4;
  const int waveId  = blockIdx.x * 8 + (tid >> 5);
  const int rowTile = waveId >> 2;
  const int colGrp  = waveId & 3;
  if (rowTile * 16 >= rows) return;
  const int row = rowTile * 16 + lo;

  v8f acc[4] = {};
#pragma unroll
  for (int kt = 0; kt < 8; ++kt) {
    v16h a = load_a16(A + (size_t)row * 256 + kt * 32 + hi * 8);
#pragma unroll
    for (int nt = 0; nt < 4; ++nt) {
      int n = colGrp * 64 + nt * 16 + lo;
      v16h b = load_b16(WT + (size_t)n * 256 + kt * 32 + hi * 8);
      acc[nt] = wmma_f16(a, b, acc[nt]);
    }
  }
#pragma unroll
  for (int nt = 0; nt < 4; ++nt) {
    int col = colGrp * 64 + nt * 16 + lo;
    float bias = b3[col];
#pragma unroll
    for (int r = 0; r < 8; ++r) {
      int orow = rowTile * 16 + r + hi * 8;
      out[(size_t)orow * 256 + col] = acc[nt][r] + bias;
    }
  }
}

// ---------------------------------------------------------------------------
extern "C" void kernel_launch(void* const* d_in, const int* in_sizes, int n_in,
                              void* d_out, int out_size, void* d_ws, size_t ws_size,
                              hipStream_t stream) {
  (void)in_sizes; (void)n_in; (void)out_size; (void)ws_size;
  const float* pts = (const float*)d_in[0];

  MlpParams P1, P2;
  for (int l = 0; l < 3; ++l) {
    int o1 = 1 + l * 6, o2 = 19 + l * 6;
    P1.W[l]  = (const float*)d_in[o1 + 0];
    P1.b[l]  = (const float*)d_in[o1 + 1];
    P1.g[l]  = (const float*)d_in[o1 + 2];
    P1.be[l] = (const float*)d_in[o1 + 3];
    P1.rm[l] = (const float*)d_in[o1 + 4];
    P1.rv[l] = (const float*)d_in[o1 + 5];
    P2.W[l]  = (const float*)d_in[o2 + 0];
    P2.b[l]  = (const float*)d_in[o2 + 1];
    P2.g[l]  = (const float*)d_in[o2 + 2];
    P2.be[l] = (const float*)d_in[o2 + 3];
    P2.rm[l] = (const float*)d_in[o2 + 4];
    P2.rv[l] = (const float*)d_in[o2 + 5];
  }
  const float* w3 = (const float*)d_in[37];
  const float* b3 = (const float*)d_in[38];

  _Float16* fbuf = (_Float16*)d_ws;                                  // [16384][256] f16
  _Float16* WT   = (_Float16*)((char*)d_ws + (size_t)16384 * 256 * 2); // [256][256] f16
  float*    out  = (float*)d_out;

  pointnet_branch_kernel<32><<<512, 128, 0, stream>>>(pts, P1, 0.1f * 0.1f, fbuf, 0);
  pointnet_branch_kernel<64><<<512, 128, 0, stream>>>(pts, P2, 0.2f * 0.2f, fbuf, 128);
  transpose_w3_kernel<<<256, 256, 0, stream>>>(w3, WT);
  final_gemm_kernel<<<512, 256, 0, stream>>>(fbuf, WT, b3, out, 16384);
}
